// PureSCN_53489522704843
// MI455X (gfx1250) — matrix operand order repaired and checked by reading, not compiled, not run on hardware
//
#include <hip/hip_runtime.h>
#include <hip/hip_bf16.h>

typedef _Float16 h16;
typedef __attribute__((ext_vector_type(16))) _Float16 v16h;
typedef __attribute__((ext_vector_type(8)))  float    v8f;
typedef unsigned int u32;

#define B_   128
#define T_   26
#define TD   25
#define V_   10000
#define E_   512
#define H_   512
#define S_   1000
#define ENC_ 2048
#define XDIM 2560   // E_+ENC_
#define G4H  2048   // 4*H_

// CDNA5 async global->LDS staging (ASYNCcnt-tracked DMA into LDS).
__device__ __forceinline__ void async_b128(u32 ldsAddr, const void* g) {
  asm volatile("global_load_async_to_lds_b128 %0, %1, off"
               :: "v"(ldsAddr), "v"(g) : "memory");
}
__device__ __forceinline__ void wait_async0() {
  asm volatile("s_wait_asynccnt 0x0" ::: "memory");
}

// ---------------------------------------------------------------------------
// Generic WMMA GEMM:  C[M,N] = A[M,K] * B[N,K]^T (+ A2*B2^T) with fused epilogue
// A,B f16 row-major (K contiguous, pitches 16B aligned, K % 8 == 0).
// Block tile 32M x 64N, 4 waves; each wave: 1 B fragment, 2 A fragments,
// 2 accumulators -> 2 v_wmma per K-slab.
// Interior blocks with K%32==0 take a software-pipelined path: LDS double
// buffer + global_load_async_to_lds_b128 prefetch, one barrier per slab.
// ---------------------------------------------------------------------------
struct GemmParams {
  const h16* A;  int lda;
  const h16* A2; int lda2;       // optional second accumulation source
  const h16* B;  int ldb;
  const h16* B2; int ldb2;
  int M, N, K, K2;
  float* Cf; int ldc;            // optional f32 output
  h16*  Ch; int ldch;            // optional f16 output
  const float* bias;             // [N] or null
  const float* addv; int ldadd;  // addv[(row%rowMod)*ldadd+col] or null
  const float* mulv; int ldmul;  // mulv[(row%rowMod)*ldmul+col] or null
  int rowMod;
  int scatter;                   // 1: predictions scatter with length mask
  const int* declen;             // [B_] decode lengths (scatter mode)
  float* outp;                   // scatter destination (B,TD,V)
};

__global__ __launch_bounds__(128) void k_gemm(GemmParams p) {
  __shared__ u32 sA[2][32][16];   // per buffer: 32 rows x 32 halves
  __shared__ u32 sB[2][64][16];   // per buffer: 64 rows x 32 halves
  const int tid  = threadIdx.x;
  const int wave = tid >> 5;
  const int lane = tid & 31;
  const int lm   = lane & 15;
  const int hi   = lane >> 4;
  const int mBase = blockIdx.y * 32;
  const int nBase = blockIdx.x * 64;
  const int nTile = nBase + wave * 16;

  // fixed staging slots: A = 128 uint4 (1/thread), B = 256 uint4 (2/thread)
  const int ar = tid >> 2, aq = tid & 3;        // A row / 8-half group
  const int br = tid >> 2, bq = tid & 3;        // B rows br and br+32
  const u32 aAddr[2]  = {(u32)(size_t)&sA[0][ar][aq * 4],
                         (u32)(size_t)&sA[1][ar][aq * 4]};
  const u32 b0Addr[2] = {(u32)(size_t)&sB[0][br][bq * 4],
                         (u32)(size_t)&sB[1][br][bq * 4]};
  const u32 b1Addr[2] = {(u32)(size_t)&sB[0][br + 32][bq * 4],
                         (u32)(size_t)&sB[1][br + 32][bq * 4]};

  v8f acc0 = {};
  v8f acc1 = {};
  const bool nfull = (nBase + 64 <= p.N);

  for (int s = 0; s < 2; ++s) {
    const h16* Ap = s ? p.A2 : p.A;
    const h16* Bp = s ? p.B2 : p.B;
    const int lda = s ? p.lda2 : p.lda;
    const int ldb = s ? p.ldb2 : p.ldb;
    const int Kk  = s ? p.K2  : p.K;
    if (!Ap) continue;                          // uniform across grid

    if (nfull && (Kk & 31) == 0) {
      // ---- pipelined path: async DMA into LDS, double buffer, 1 barrier/slab
      const int nslab = Kk >> 5;
      const h16* Aptr  = Ap + (long)(mBase + ar) * lda + aq * 8;
      const h16* Bptr0 = Bp + (long)(nBase + br) * ldb + bq * 8;
      const h16* Bptr1 = Bp + (long)(nBase + br + 32) * ldb + bq * 8;
      int cur = 0;
      async_b128(aAddr[0], Aptr);
      async_b128(b0Addr[0], Bptr0);
      async_b128(b1Addr[0], Bptr1);
      for (int i = 0; i < nslab; ++i) {
        wait_async0();          // my slab-i transfers landed in LDS
        __syncthreads();        // everyone's landed; prev-prev reads all done
        int nb = cur ^ 1;
        if (i + 1 < nslab) {    // uniform
          Aptr += 32; Bptr0 += 32; Bptr1 += 32;
          async_b128(aAddr[nb], Aptr);
          async_b128(b0Addr[nb], Bptr0);
          async_b128(b1Addr[nb], Bptr1);
        }
        union { v16h v; u32 u[8]; } bf, a0, a1;
#pragma unroll
        for (int j = 0; j < 8; ++j)
          bf.u[j] = sB[cur][wave * 16 + lm][j + 8 * hi];
#pragma unroll
        for (int j = 0; j < 8; ++j) {
          int ua = j + 8 * hi;
          a0.u[j] = sA[cur][lm][ua];
          a1.u[j] = sA[cur][16 + lm][ua];
        }
        acc0 = __builtin_amdgcn_wmma_f32_16x16x32_f16(
            false, a0.v, false, bf.v, (short)0, acc0, false, false);
        acc1 = __builtin_amdgcn_wmma_f32_16x16x32_f16(
            false, a1.v, false, bf.v, (short)0, acc1, false, false);
        cur ^= 1;
      }
      __syncthreads();          // quiesce buffers before next source segment
    } else {
      // ---- guarded fallback (K tail not 32-aligned or N-edge block)
      for (int k0 = 0; k0 < Kk; k0 += 32) {
        {
          int kg = k0 + aq * 8;
          uint4 v = {0u, 0u, 0u, 0u};
          if (kg + 8 <= Kk)
            v = *reinterpret_cast<const uint4*>(Ap + (long)(mBase + ar) * lda + kg);
          *reinterpret_cast<uint4*>(&sA[0][ar][aq * 4]) = v;
        }
#pragma unroll
        for (int it = 0; it < 2; ++it) {
          int r = br + it * 32;
          int n = nBase + r;
          int kg = k0 + bq * 8;
          uint4 v = {0u, 0u, 0u, 0u};
          if (n < p.N && kg + 8 <= Kk)
            v = *reinterpret_cast<const uint4*>(Bp + (long)n * ldb + kg);
          *reinterpret_cast<uint4*>(&sB[0][r][bq * 4]) = v;
        }
        __syncthreads();
        union { v16h v; u32 u[8]; } bf, a0, a1;
#pragma unroll
        for (int j = 0; j < 8; ++j)
          bf.u[j] = sB[0][wave * 16 + lm][j + 8 * hi];
#pragma unroll
        for (int j = 0; j < 8; ++j) {
          int ua = j + 8 * hi;
          a0.u[j] = sA[0][lm][ua];
          a1.u[j] = sA[0][16 + lm][ua];
        }
        acc0 = __builtin_amdgcn_wmma_f32_16x16x32_f16(
            false, a0.v, false, bf.v, (short)0, acc0, false, false);
        acc1 = __builtin_amdgcn_wmma_f32_16x16x32_f16(
            false, a1.v, false, bf.v, (short)0, acc1, false, false);
        __syncthreads();
      }
    }
  }

  // Epilogue. C layout: lane -> n=lane%16, VGPR r -> m=r+8*(lane/16).
  const int col = nTile + lm;
  if (col >= p.N) return;
#pragma unroll
  for (int mi = 0; mi < 2; ++mi) {
#pragma unroll
    for (int r = 0; r < 8; ++r) {
      int row = mBase + mi * 16 + r + 8 * hi;
      float v = mi ? acc1[r] : acc0[r];
      if (p.bias) v += p.bias[col];
      if (p.addv || p.mulv) {
        int rb = row % p.rowMod;
        if (p.addv) v += p.addv[(long)rb * p.ldadd + col];
        if (p.mulv) v *= p.mulv[(long)rb * p.ldmul + col];
      }
      if (p.scatter) {
        int t = row >> 7, b = row & 127;          // row = t*128 + b
        float mk = (t < p.declen[b]) ? 1.f : 0.f;
        p.outp[((long)b * TD + t) * V_ + col] = v * mk;
      } else {
        if (p.Cf) p.Cf[(long)row * p.ldc  + col] = v;
        if (p.Ch) p.Ch[(long)row * p.ldch + col] = (h16)v;
      }
    }
  }
}

// ---------------------------------------------------------------------------
// Sort (stable descending by length), gather + convert, elementwise kernels
// ---------------------------------------------------------------------------
__global__ __launch_bounds__(128) void k_sort(
    const int* cap_len, const int* caps,
    int* sort_ind, int* declen_i, int* caps_s,
    float* out_caps, float* out_declen, float* out_sortind) {
  __shared__ int len[B_];
  __shared__ int sind[B_];
  int i = threadIdx.x;
  len[i] = cap_len[i];
  __syncthreads();
  int li = len[i], rank = 0;
  for (int j = 0; j < B_; ++j) {
    int lj = len[j];
    rank += (lj > li) || (lj == li && j < i);
  }
  sind[rank] = i;
  __syncthreads();
  int src = sind[i];
  sort_ind[i] = src;
  out_sortind[i] = (float)src;
  int L = len[src];
  declen_i[i] = L - 1;
  out_declen[i] = (float)(L - 1);
  for (int t = 0; t < T_; ++t) {
    int cv = caps[src * T_ + t];
    caps_s[i * T_ + t] = cv;
    out_caps[i * T_ + t] = (float)cv;
  }
}

__global__ __launch_bounds__(256) void k_gather_enc(
    const float* enc, const int* sind, float* enc_s, h16* enc_s_h) {
  int idx = blockIdx.x * 256 + threadIdx.x;
  if (idx >= B_ * ENC_) return;
  int b = idx >> 11, e = idx & 2047;
  float v = enc[(long)sind[b] * ENC_ + e];
  enc_s[idx] = v;
  enc_s_h[idx] = (h16)v;
}

__global__ __launch_bounds__(256) void k_gather_tag(
    const float* tag, const int* sind, h16* tag_s_h) {
  int idx = blockIdx.x * 256 + threadIdx.x;
  if (idx >= B_ * S_) return;
  int b = idx / S_, s = idx % S_;
  tag_s_h[idx] = (h16)tag[(long)sind[b] * S_ + s];
}

__global__ __launch_bounds__(256) void k_gather_emb(
    const float* emb, const int* caps_s, h16* Xemb_h) {
  int idx = blockIdx.x * 256 + threadIdx.x;
  if (idx >= TD * B_ * E_) return;
  int e = idx & 511, r = idx >> 9;
  int t = r >> 7, b = r & 127;
  int tok = caps_s[b * T_ + t];
  Xemb_h[idx] = (h16)emb[(long)tok * E_ + e];
}

__global__ __launch_bounds__(256) void k_cvt(const float* s, h16* d, int n) {
  int idx = blockIdx.x * 256 + threadIdx.x;
  if (idx < n) d[idx] = (h16)s[idx];
}

__global__ __launch_bounds__(256) void k_bias2(
    const float* bi, const float* bh, float* b2) {
  int idx = blockIdx.x * 256 + threadIdx.x;
  if (idx < G4H) b2[idx] = bi[idx] + bh[idx];
}

__device__ __forceinline__ float sigm(float x) { return 1.f / (1.f + __expf(-x)); }

__global__ __launch_bounds__(256) void k_lstm(
    const float* gates, float* c, float* h, h16* h_h, h16* hstore,
    const int* declen_i, int t) {
  int idx = blockIdx.x * 256 + threadIdx.x;
  if (idx >= B_ * H_) return;
  int b = idx >> 9, j = idx & 511;
  const float* gr = gates + (long)b * G4H;
  float gi = sigm(gr[j]);
  float gf = sigm(gr[H_ + j]);
  float gg = tanhf(gr[2 * H_ + j]);
  float go = sigm(gr[3 * H_ + j]);
  float cn = gf * c[idx] + gi * gg;
  float hn = go * tanhf(cn);
  bool mk = t < declen_i[b];
  float hm = mk ? hn : h[idx];
  float cm = mk ? cn : c[idx];
  h[idx] = hm; c[idx] = cm;
  h_h[idx] = (h16)hm;
  hstore[(long)(t * B_ + b) * H_ + j] = (h16)hn;
}

// ---------------------------------------------------------------------------
extern "C" void kernel_launch(void* const* d_in, const int* in_sizes, int n_in,
                              void* d_out, int out_size, void* d_ws, size_t ws_size,
                              hipStream_t stream) {
  (void)in_sizes; (void)n_in; (void)out_size; (void)ws_size;
  const float* enc_in  = (const float*)d_in[0];
  const float* tag_in  = (const float*)d_in[1];
  const int*   caps_in = (const int*)d_in[2];
  const int*   len_in  = (const int*)d_in[3];
  const float* emb     = (const float*)d_in[4];
  const float* W_ia    = (const float*)d_in[5];
  const float* W_ib    = (const float*)d_in[6];
  const float* W_ic    = (const float*)d_in[7];
  const float* W_ha    = (const float*)d_in[8];
  const float* W_hb    = (const float*)d_in[9];
  const float* W_hc    = (const float*)d_in[10];
  const float* b_ih    = (const float*)d_in[11];
  const float* b_hh    = (const float*)d_in[12];
  const float* W_init_h= (const float*)d_in[13];
  const float* b_init_h= (const float*)d_in[14];
  const float* W_init_c= (const float*)d_in[15];
  const float* b_init_c= (const float*)d_in[16];
  const float* W_fc    = (const float*)d_in[17];
  const float* b_fc    = (const float*)d_in[18];

  float* out = (float*)d_out;
  float* out_preds   = out;                              // B*TD*V
  float* out_caps    = out + (long)B_ * TD * V_;         // B*T
  float* out_declen  = out_caps + B_ * T_;               // B
  float* out_sortind = out_declen + B_;                  // B

  // workspace bump allocator (256B aligned)
  size_t off = 0;
  auto alloc = [&](size_t bytes) -> void* {
    void* p = (char*)d_ws + off;
    off = (off + bytes + 255) & ~(size_t)255;
    return p;
  };
  int*   sort_ind = (int*)  alloc(B_ * 4);
  int*   declen_i = (int*)  alloc(B_ * 4);
  int*   caps_s   = (int*)  alloc(B_ * T_ * 4);
  float* enc_s    = (float*)alloc((size_t)B_ * ENC_ * 4);
  h16*   enc_s_h  = (h16*)  alloc((size_t)B_ * ENC_ * 2);
  h16*   tag_s_h  = (h16*)  alloc((size_t)B_ * S_ * 2);
  h16*   Xemb_h   = (h16*)  alloc((size_t)TD * B_ * E_ * 2);
  h16*   Wia_h    = (h16*)  alloc((size_t)G4H * XDIM * 2);
  h16*   Wib_h    = (h16*)  alloc((size_t)G4H * S_ * 2);
  h16*   Wic_h    = (h16*)  alloc((size_t)G4H * H_ * 2);
  h16*   Wha_h    = (h16*)  alloc((size_t)G4H * H_ * 2);
  h16*   Whb_h    = (h16*)  alloc((size_t)G4H * S_ * 2);
  h16*   Whc_h    = (h16*)  alloc((size_t)G4H * H_ * 2);
  h16*   Winh_h   = (h16*)  alloc((size_t)H_ * ENC_ * 2);
  h16*   Winc_h   = (h16*)  alloc((size_t)H_ * ENC_ * 2);
  h16*   Wfc_h    = (h16*)  alloc((size_t)V_ * H_ * 2);
  float* sb_i     = (float*)alloc((size_t)B_ * G4H * 4);
  float* sb_h     = (float*)alloc((size_t)B_ * G4H * 4);
  float* xa_enc   = (float*)alloc((size_t)B_ * G4H * 4);
  h16*   XA_h     = (h16*)  alloc((size_t)TD * B_ * G4H * 2);
  h16*   HA_h     = (h16*)  alloc((size_t)B_ * G4H * 2);
  float* gates    = (float*)alloc((size_t)B_ * G4H * 4);
  float* hbuf     = (float*)alloc((size_t)B_ * H_ * 4);
  float* cbuf     = (float*)alloc((size_t)B_ * H_ * 4);
  h16*   h_h      = (h16*)  alloc((size_t)B_ * H_ * 2);
  h16*   hstore   = (h16*)  alloc((size_t)TD * B_ * H_ * 2);
  float* bias2    = (float*)alloc((size_t)G4H * 4);

  auto gemm = [&](GemmParams p) {
    dim3 grid((p.N + 63) / 64, p.M / 32);
    k_gemm<<<grid, 128, 0, stream>>>(p);
  };
  auto base = [&]() {
    GemmParams p{};
    p.rowMod = 1;
    return p;
  };

  // 1) sort + caption gather + small outputs
  k_sort<<<1, 128, 0, stream>>>(len_in, caps_in, sort_ind, declen_i, caps_s,
                                out_caps, out_declen, out_sortind);
  // 2) gathers / conversions
  k_gather_enc<<<(B_ * ENC_ + 255) / 256, 256, 0, stream>>>(enc_in, sort_ind, enc_s, enc_s_h);
  k_gather_tag<<<(B_ * S_ + 255) / 256, 256, 0, stream>>>(tag_in, sort_ind, tag_s_h);
  k_gather_emb<<<(TD * B_ * E_ + 255) / 256, 256, 0, stream>>>(emb, caps_s, Xemb_h);
  auto cvt = [&](const float* s, h16* d, long n) {
    k_cvt<<<(unsigned)((n + 255) / 256), 256, 0, stream>>>(s, d, (int)n);
  };
  cvt(W_ia, Wia_h, (long)G4H * XDIM);
  cvt(W_ib, Wib_h, (long)G4H * S_);
  cvt(W_ic, Wic_h, (long)G4H * H_);
  cvt(W_ha, Wha_h, (long)G4H * H_);
  cvt(W_hb, Whb_h, (long)G4H * S_);
  cvt(W_hc, Whc_h, (long)G4H * H_);
  cvt(W_init_h, Winh_h, (long)H_ * ENC_);
  cvt(W_init_c, Winc_h, (long)H_ * ENC_);
  cvt(W_fc, Wfc_h, (long)V_ * H_);
  k_bias2<<<G4H / 256, 256, 0, stream>>>(b_ih, b_hh, bias2);

  // 3) time-invariant GEMMs
  { GemmParams p = base();   // sb_i = tag_s @ W_ib^T
    p.A = tag_s_h; p.lda = S_; p.B = Wib_h; p.ldb = S_;
    p.M = B_; p.N = G4H; p.K = S_; p.Cf = sb_i; p.ldc = G4H; gemm(p); }
  { GemmParams p = base();   // sb_h = tag_s @ W_hb^T
    p.A = tag_s_h; p.lda = S_; p.B = Whb_h; p.ldb = S_;
    p.M = B_; p.N = G4H; p.K = S_; p.Cf = sb_h; p.ldc = G4H; gemm(p); }
  { GemmParams p = base();   // h0 = enc_s @ W_init_h^T + b  (f32 + f16 copies)
    p.A = enc_s_h; p.lda = ENC_; p.B = Winh_h; p.ldb = ENC_;
    p.M = B_; p.N = H_; p.K = ENC_; p.bias = b_init_h;
    p.Cf = hbuf; p.ldc = H_; p.Ch = h_h; p.ldch = H_; gemm(p); }
  { GemmParams p = base();   // c0
    p.A = enc_s_h; p.lda = ENC_; p.B = Winc_h; p.ldb = ENC_;
    p.M = B_; p.N = H_; p.K = ENC_; p.bias = b_init_c;
    p.Cf = cbuf; p.ldc = H_; gemm(p); }
  { GemmParams p = base();   // xa_enc = enc_s @ W_ia[:,:,E:]^T  (time-invariant)
    p.A = enc_s_h; p.lda = ENC_; p.B = Wia_h + E_; p.ldb = XDIM;
    p.M = B_; p.N = G4H; p.K = ENC_; p.Cf = xa_enc; p.ldc = G4H; gemm(p); }
  { GemmParams p = base();   // XA[t,b] = (emb_t @ W_ia[:,:,:E]^T + xa_enc) * sb_i  -> f16
    p.A = Xemb_h; p.lda = E_; p.B = Wia_h; p.ldb = XDIM;
    p.M = TD * B_; p.N = G4H; p.K = E_;
    p.addv = xa_enc; p.ldadd = G4H; p.mulv = sb_i; p.ldmul = G4H; p.rowMod = B_;
    p.Ch = XA_h; p.ldch = G4H; gemm(p); }

  // 4) recurrent loop: HA gemm -> 4 dual-source gate gemms -> LSTM elementwise
  for (int t = 0; t < TD; ++t) {
    { GemmParams p = base(); // HA = (h @ W_ha^T) * sb_h -> f16
      p.A = h_h; p.lda = H_; p.B = Wha_h; p.ldb = H_;
      p.M = B_; p.N = G4H; p.K = H_;
      p.mulv = sb_h; p.ldmul = G4H; p.rowMod = B_;
      p.Ch = HA_h; p.ldch = G4H; gemm(p); }
    for (int g = 0; g < 4; ++g) {
      GemmParams p = base(); // gates_g = XA_g @ W_ic_g^T + HA_g @ W_hc_g^T + bias
      p.A  = XA_h + (size_t)t * B_ * G4H + g * H_; p.lda  = G4H;
      p.A2 = HA_h + g * H_;                        p.lda2 = G4H;
      p.B  = Wic_h + (size_t)g * H_ * H_;          p.ldb  = H_;
      p.B2 = Whc_h + (size_t)g * H_ * H_;          p.ldb2 = H_;
      p.M = B_; p.N = H_; p.K = H_; p.K2 = H_;
      p.bias = bias2 + g * H_;
      p.Cf = gates + g * H_; p.ldc = G4H; gemm(p);
    }
    k_lstm<<<(B_ * H_) / 256, 256, 0, stream>>>(gates, cbuf, hbuf, h_h, hstore,
                                                declen_i, t);
  }

  // 5) one big logits GEMM with fused length-mask + (B,T,V) transpose scatter
  { GemmParams p = base();
    p.A = hstore; p.lda = H_; p.B = Wfc_h; p.ldb = H_;
    p.M = TD * B_; p.N = V_; p.K = H_; p.bias = b_fc;
    p.scatter = 1; p.declen = declen_i; p.outp = out_preds; gemm(p); }
}
